// EnhancedGraphTransformerLayer_40132174414149
// MI455X (gfx1250) — compile-verified, matrix operand order checked by reading, not run on hardware
//
#include <hip/hip_runtime.h>
#include <hip/hip_bf16.h>
#include <stdint.h>

// ---------- types ----------
typedef __bf16 bf16;
typedef __bf16 bf16x8 __attribute__((ext_vector_type(8)));
typedef __bf16 v16bf  __attribute__((ext_vector_type(16)));
typedef float  v8f    __attribute__((ext_vector_type(8)));
typedef float  f32x4  __attribute__((ext_vector_type(4)));

#define N_NODES 65536
#define EMBED   512
#define FF      2048
#define NHEADS  8
#define HDIM    64

// LDS stage: 128 rows x 64 K (bf16), row pitch 72 elems (144B).
// Row-strided b128 reads hit bank 36*L mod 64 -> permutation over 16 lanes.
#define TROWS   128
#define TK      64
#define TPITCH  72

// ---------- wave32 reduction via ds_swizzle (no wave64 assumptions) ----------
template<int IMM>
__device__ inline float swz_add(float x) {
    return x + __int_as_float(__builtin_amdgcn_ds_swizzle(__float_as_int(x), IMM));
}
__device__ inline float wave_sum(float x) {
    x = swz_add<(1  << 10) | 31>(x);
    x = swz_add<(2  << 10) | 31>(x);
    x = swz_add<(4  << 10) | 31>(x);
    x = swz_add<(8  << 10) | 31>(x);
    x = swz_add<(16 << 10) | 31>(x);
    return x;
}

// ---------- CDNA5 async global->LDS copy (ASYNCcnt-tracked, no VGPR round trip) ----------
__device__ inline void async_copy16(uint32_t lds_addr, const bf16* g) {
    asm volatile("global_load_async_to_lds_b128 %0, %1, off"
                 :: "v"(lds_addr), "v"((uint64_t)(uintptr_t)g)
                 : "memory");
}
// 64B of one row per operand per thread: 4 consecutive 16B chunks.
__device__ inline void async_copy64(uint32_t lds_addr, const bf16* g) {
    async_copy16(lds_addr,      g);
    async_copy16(lds_addr + 16, g + 8);
    async_copy16(lds_addr + 32, g + 16);
    async_copy16(lds_addr + 48, g + 24);
}
__device__ inline void wait_async_le8() {
    asm volatile("s_wait_asynccnt 0x8" ::: "memory");
}
__device__ inline void wait_async_0() {
    asm volatile("s_wait_asynccnt 0x0" ::: "memory");
}

// ---------- fp32 -> bf16 convert ----------
__global__ __launch_bounds__(256) void cvt_kernel(const float* __restrict__ src,
                                                  bf16* __restrict__ dst, int n) {
    int i = blockIdx.x * 256 + threadIdx.x;
    if (i < n) dst[i] = (bf16)src[i];
}

// ---------- LayerNorm: fp32 row(512) -> bf16 row ----------
__global__ __launch_bounds__(256) void ln_kernel(const float* __restrict__ x,
                                                 const float* __restrict__ g,
                                                 const float* __restrict__ b,
                                                 bf16* __restrict__ out) {
    int wave = threadIdx.x >> 5, lane = threadIdx.x & 31;
    int row = blockIdx.x * 8 + wave;
    const float* xr = x + (size_t)row * EMBED + lane * 16;
    float v[16];
#pragma unroll
    for (int i = 0; i < 4; ++i) {
        f32x4 t = *reinterpret_cast<const f32x4*>(xr + i * 4);
#pragma unroll
        for (int j = 0; j < 4; ++j) v[i * 4 + j] = t[j];
    }
    float s = 0.f, q = 0.f;
#pragma unroll
    for (int i = 0; i < 16; ++i) { s += v[i]; q += v[i] * v[i]; }
    s = wave_sum(s); q = wave_sum(q);
    float mean = s * (1.f / EMBED);
    float var  = q * (1.f / EMBED) - mean * mean;
    float rs   = rsqrtf(var + 1e-5f);
    int c0 = lane * 16;
    bf16x8 o0, o1;
#pragma unroll
    for (int i = 0; i < 8; ++i) {
        o0[i] = (bf16)((v[i]     - mean) * rs * g[c0 + i]     + b[c0 + i]);
        o1[i] = (bf16)((v[i + 8] - mean) * rs * g[c0 + i + 8] + b[c0 + i + 8]);
    }
    bf16* orow = out + (size_t)row * EMBED + c0;
    *reinterpret_cast<bf16x8*>(orow)     = o0;
    *reinterpret_cast<bf16x8*>(orow + 8) = o1;
}

// 16-bit WMMA operand lane layout (ISA 7.12.2): lanes 0-15 hold K 0-7 & 16-23,
// lanes 16-31 hold K 8-15 & 24-31 -> two 16B chunks at +0/+32B from (row, koff).
__device__ inline v16bf load_tile16(const bf16* base) {
    bf16x8 lo = *reinterpret_cast<const bf16x8*>(base);
    bf16x8 hi = *reinterpret_cast<const bf16x8*>(base + 16);
    return __builtin_shufflevector(lo, hi, 0, 1, 2, 3, 4, 5, 6, 7,
                                           8, 9, 10, 11, 12, 13, 14, 15);
}

// ---------- WMMA bf16 GEMM, double-buffered async-LDS pipeline ----------
// out = A(MxK) * W(NcxK)^T + bias [+res][relu].  Block = 256 thr = 8 waves,
// block tile 128x128, wave tile 32x64, staged K-slice 64 (2 wmma K-steps).
template<int RELU, int RES, int OUTF32>
__global__ __launch_bounds__(256) void gemm_bf16_kernel(
    const bf16* __restrict__ A, const bf16* __restrict__ W,
    const float* __restrict__ bias, const float* __restrict__ res,
    float* __restrict__ outF, bf16* __restrict__ outB,
    int M, int K, int Nc) {
    __shared__ __align__(16) bf16 sA[2][TROWS * TPITCH];
    __shared__ __align__(16) bf16 sW[2][TROWS * TPITCH];

    const int tid  = threadIdx.x;
    const int wave = tid >> 5, lane = tid & 31;
    const int m0 = blockIdx.y * 128;
    const int n0 = blockIdx.x * 128;

    // ---- loader mapping: 2 threads per row, each stages 64B (4 chunks) ----
    const int lrow = tid >> 1;                 // 0..127
    const int lhalf = (tid & 1) * 32;          // K-element offset 0 / 32
    const bf16* gA = A + (size_t)(m0 + lrow) * K + lhalf;
    const bf16* gW = W + (size_t)(n0 + lrow) * K + lhalf;
    const uint32_t ldsOff = (uint32_t)(lrow * TPITCH + lhalf) * 2;   // bytes
    const uint32_t lA[2] = { (uint32_t)(uintptr_t)&sA[0][0] + ldsOff,
                             (uint32_t)(uintptr_t)&sA[1][0] + ldsOff };
    const uint32_t lW[2] = { (uint32_t)(uintptr_t)&sW[0][0] + ldsOff,
                             (uint32_t)(uintptr_t)&sW[1][0] + ldsOff };

    // ---- compute mapping ----
    const int lr   = lane & 15;
    const int koff = (lane >> 4) * 8;
    const int aoff0 = ((wave >> 1) * 32 + lr) * TPITCH + koff;       // elems
    const int aoff1 = aoff0 + 16 * TPITCH;
    int woff[4];
#pragma unroll
    for (int t = 0; t < 4; ++t)
        woff[t] = ((wave & 1) * 64 + t * 16 + lr) * TPITCH + koff;

    v8f c[2][4] = {};
    const int nk = K / TK;

    // prologue: stage buffer 0
    async_copy64(lA[0], gA);
    async_copy64(lW[0], gW);

    auto compute = [&](int cur) {
        const bf16* bA = &sA[cur][0];
        const bf16* bW = &sW[cur][0];
#pragma unroll
        for (int ks = 0; ks < 2; ++ks) {       // two 16x16x32 K-steps per stage
            v16bf a0 = load_tile16(bA + aoff0 + ks * 32);
            v16bf a1 = load_tile16(bA + aoff1 + ks * 32);
#pragma unroll
            for (int t = 0; t < 4; ++t) {
                v16bf bt = load_tile16(bW + woff[t] + ks * 32);
                c[0][t] = __builtin_amdgcn_wmma_f32_16x16x32_bf16(
                    false, a0, false, bt, (short)0, c[0][t], false, false);
                c[1][t] = __builtin_amdgcn_wmma_f32_16x16x32_bf16(
                    false, a1, false, bt, (short)0, c[1][t], false, false);
            }
        }
    };

    // steady state: branchless issue -> split-wait -> barrier -> 16 WMMAs
    for (int ki = 0; ki < nk - 1; ++ki) {
        const int cur = ki & 1;
        __syncthreads();                       // prev compute done: safe to overwrite cur^1
        async_copy64(lA[cur ^ 1], gA + (ki + 1) * TK);
        async_copy64(lW[cur ^ 1], gW + (ki + 1) * TK);
        wait_async_le8();                      // cur landed; next stays in flight
        __syncthreads();                       // all waves' cur data visible
        compute(cur);
    }
    // epilogue stage
    wait_async_0();
    __syncthreads();
    compute((nk - 1) & 1);

    // C layout (ISA 7.12.2): VGPR r, lanes 0-15 -> M=r, lanes 16-31 -> M=r+8; N=lane&15
    const int wm0 = m0 + (wave >> 1) * 32;
    const int wn0 = n0 + (wave & 1) * 64;
    const int colb   = lane & 15;
    const int rshift = (lane >> 4) * 8;
#pragma unroll
    for (int s = 0; s < 2; ++s) {
#pragma unroll
        for (int t = 0; t < 4; ++t) {
            int col  = wn0 + t * 16 + colb;
            int rowb = wm0 + s * 16 + rshift;
            float bv = bias[col];
#pragma unroll
            for (int r = 0; r < 8; ++r) {
                int row = rowb + r;
                float val = c[s][t][r] + bv;
                if (RES)  val += res[(size_t)row * Nc + col];
                if (RELU) val = fmaxf(val, 0.f);
                if (OUTF32) outF[(size_t)row * Nc + col] = val;
                else        outB[(size_t)row * Nc + col] = (bf16)val;
            }
        }
    }
}

// ---------- per-node multi-head attention (8 heads x 64 dim, per-node softmax) ----------
__global__ __launch_bounds__(256) void attn_kernel(
    const bf16* __restrict__ q, const bf16* __restrict__ k,
    const bf16* __restrict__ v, bf16* __restrict__ out) {
    __shared__ __align__(16) bf16 sQ[8][512];
    __shared__ __align__(16) bf16 sK[8][512];
    __shared__ __align__(16) bf16 sV[8][512];
    __shared__ float sS[8][64];

    int wave = threadIdx.x >> 5, lane = threadIdx.x & 31;
    int node = blockIdx.x * 8 + wave;
    size_t base = (size_t)node * EMBED;
    int c0 = lane * 16;

    *reinterpret_cast<bf16x8*>(&sQ[wave][c0])     = *reinterpret_cast<const bf16x8*>(q + base + c0);
    *reinterpret_cast<bf16x8*>(&sQ[wave][c0 + 8]) = *reinterpret_cast<const bf16x8*>(q + base + c0 + 8);
    *reinterpret_cast<bf16x8*>(&sK[wave][c0])     = *reinterpret_cast<const bf16x8*>(k + base + c0);
    *reinterpret_cast<bf16x8*>(&sK[wave][c0 + 8]) = *reinterpret_cast<const bf16x8*>(k + base + c0 + 8);
    *reinterpret_cast<bf16x8*>(&sV[wave][c0])     = *reinterpret_cast<const bf16x8*>(v + base + c0);
    *reinterpret_cast<bf16x8*>(&sV[wave][c0 + 8]) = *reinterpret_cast<const bf16x8*>(v + base + c0 + 8);
    __syncthreads();

    int h = lane >> 2;
    {
        int g0 = lane & 3;
        float acc0 = 0.f, acc1 = 0.f;
#pragma unroll
        for (int d = 0; d < HDIM; d += 8) {
            bf16x8 qv  = *reinterpret_cast<const bf16x8*>(&sQ[wave][h * HDIM + d]);
            bf16x8 kv0 = *reinterpret_cast<const bf16x8*>(&sK[wave][g0 * HDIM + d]);
            bf16x8 kv1 = *reinterpret_cast<const bf16x8*>(&sK[wave][(g0 + 4) * HDIM + d]);
#pragma unroll
            for (int i = 0; i < 8; ++i) {
                float qf = (float)qv[i];
                acc0 += qf * (float)kv0[i];
                acc1 += qf * (float)kv1[i];
            }
        }
        sS[wave][h * 8 + g0]     = acc0 * 0.125f;   // 1/sqrt(64)
        sS[wave][h * 8 + g0 + 4] = acc1 * 0.125f;
    }
    __syncthreads();

    if (threadIdx.x < 64) {
        int nl = threadIdx.x >> 3, hh = threadIdx.x & 7;
        float* s = &sS[nl][hh * 8];
        float mx = s[0];
#pragma unroll
        for (int i = 1; i < 8; ++i) mx = fmaxf(mx, s[i]);
        float sum = 0.f;
#pragma unroll
        for (int i = 0; i < 8; ++i) { float e = __expf(s[i] - mx); s[i] = e; sum += e; }
        float inv = 1.f / sum;
#pragma unroll
        for (int i = 0; i < 8; ++i) s[i] *= inv;
    }
    __syncthreads();

    int d0 = (lane & 3) * 16;
    float o[16] = {};
#pragma unroll
    for (int g = 0; g < 8; ++g) {
        float aw = sS[wave][h * 8 + g];
        bf16x8 v0 = *reinterpret_cast<const bf16x8*>(&sV[wave][g * HDIM + d0]);
        bf16x8 v1 = *reinterpret_cast<const bf16x8*>(&sV[wave][g * HDIM + d0 + 8]);
#pragma unroll
        for (int i = 0; i < 8; ++i) {
            o[i]     += aw * (float)v0[i];
            o[i + 8] += aw * (float)v1[i];
        }
    }
    bf16x8 o0, o1;
#pragma unroll
    for (int i = 0; i < 8; ++i) { o0[i] = (bf16)o[i]; o1[i] = (bf16)o[i + 8]; }
    bf16* orow = out + base + h * HDIM + d0;
    *reinterpret_cast<bf16x8*>(orow)     = o0;
    *reinterpret_cast<bf16x8*>(orow + 8) = o1;
}

// ---------- host launch ----------
extern "C" void kernel_launch(void* const* d_in, const int* in_sizes, int n_in,
                              void* d_out, int out_size, void* d_ws, size_t ws_size,
                              hipStream_t stream) {
    (void)in_sizes; (void)n_in; (void)out_size; (void)ws_size;
    const float* x     = (const float*)d_in[0];
    const float* Wq    = (const float*)d_in[1];  const float* bq    = (const float*)d_in[2];
    const float* Wk    = (const float*)d_in[3];  const float* bk    = (const float*)d_in[4];
    const float* Wv    = (const float*)d_in[5];  const float* bv    = (const float*)d_in[6];
    const float* Wo    = (const float*)d_in[7];  const float* bo    = (const float*)d_in[8];
    const float* W1    = (const float*)d_in[9];  const float* b1    = (const float*)d_in[10];
    const float* W2    = (const float*)d_in[11]; const float* b2    = (const float*)d_in[12];
    const float* g1    = (const float*)d_in[13]; const float* beta1 = (const float*)d_in[14];
    const float* g2    = (const float*)d_in[15]; const float* beta2 = (const float*)d_in[16];

    char* p = (char*)d_ws;
    auto carve = [&](size_t bytes) {
        char* r = p;
        p += (bytes + 255) & ~(size_t)255;
        return r;
    };
    bf16* wq_b = (bf16*)carve((size_t)EMBED * EMBED * 2);
    bf16* wk_b = (bf16*)carve((size_t)EMBED * EMBED * 2);
    bf16* wv_b = (bf16*)carve((size_t)EMBED * EMBED * 2);
    bf16* wo_b = (bf16*)carve((size_t)EMBED * EMBED * 2);
    bf16* w1_b = (bf16*)carve((size_t)FF * EMBED * 2);
    bf16* w2_b = (bf16*)carve((size_t)FF * EMBED * 2);
    bf16* h1   = (bf16*)carve((size_t)N_NODES * EMBED * 2);        // also reused as h2
    bf16* qb   = (bf16*)carve((size_t)N_NODES * EMBED * 2);        // q | (ff1 aliases q..attn)
    bf16* kb   = (bf16*)carve((size_t)N_NODES * EMBED * 2);
    bf16* vb   = (bf16*)carve((size_t)N_NODES * EMBED * 2);
    bf16* ab   = (bf16*)carve((size_t)N_NODES * EMBED * 2);        // attention output
    bf16* ff1  = qb;                                               // N*FF == 4*N*EMBED, exact alias
    float* x1  = (float*)carve((size_t)N_NODES * EMBED * 4);

    cvt_kernel<<<EMBED * EMBED / 256, 256, 0, stream>>>(Wq, wq_b, EMBED * EMBED);
    cvt_kernel<<<EMBED * EMBED / 256, 256, 0, stream>>>(Wk, wk_b, EMBED * EMBED);
    cvt_kernel<<<EMBED * EMBED / 256, 256, 0, stream>>>(Wv, wv_b, EMBED * EMBED);
    cvt_kernel<<<EMBED * EMBED / 256, 256, 0, stream>>>(Wo, wo_b, EMBED * EMBED);
    cvt_kernel<<<FF * EMBED / 256, 256, 0, stream>>>(W1, w1_b, FF * EMBED);
    cvt_kernel<<<FF * EMBED / 256, 256, 0, stream>>>(W2, w2_b, FF * EMBED);

    ln_kernel<<<N_NODES / 8, 256, 0, stream>>>(x, g1, beta1, h1);

    dim3 g512(EMBED / 128, N_NODES / 128);
    gemm_bf16_kernel<0, 0, 0><<<g512, 256, 0, stream>>>(h1, wq_b, bq, nullptr, nullptr, qb,
                                                        N_NODES, EMBED, EMBED);
    gemm_bf16_kernel<0, 0, 0><<<g512, 256, 0, stream>>>(h1, wk_b, bk, nullptr, nullptr, kb,
                                                        N_NODES, EMBED, EMBED);
    gemm_bf16_kernel<0, 0, 0><<<g512, 256, 0, stream>>>(h1, wv_b, bv, nullptr, nullptr, vb,
                                                        N_NODES, EMBED, EMBED);

    attn_kernel<<<N_NODES / 8, 256, 0, stream>>>(qb, kb, vb, ab);

    gemm_bf16_kernel<0, 1, 1><<<g512, 256, 0, stream>>>(ab, wo_b, bo, x, x1, nullptr,
                                                        N_NODES, EMBED, EMBED);
    ln_kernel<<<N_NODES / 8, 256, 0, stream>>>(x1, g2, beta2, h1);

    dim3 gff(FF / 128, N_NODES / 128);
    gemm_bf16_kernel<1, 0, 0><<<gff, 256, 0, stream>>>(h1, w1_b, b1, nullptr, nullptr, ff1,
                                                       N_NODES, EMBED, FF);
    gemm_bf16_kernel<0, 1, 1><<<g512, 256, 0, stream>>>(ff1, w2_b, b2, x1, (float*)d_out, nullptr,
                                                        N_NODES, FF, EMBED);
}
